// TrigonModule_77103252898121
// MI455X (gfx1250) — matrix operand order repaired and checked by reading, not compiled, not run on hardware
//
#include <hip/hip_runtime.h>
#include <cstdint>

typedef _Float16 half_t;
typedef __attribute__((ext_vector_type(16))) _Float16 v16h;
typedef __attribute__((ext_vector_type(8)))  float    v8f;

#define NB   8
#define NRE  400
#define NLI  128
#define NK   64
#define NROW (NB*NRE*NK)   // 204800 rows of z (each 32 wide)

__device__ __forceinline__ float fastrcp(float x){ return __builtin_amdgcn_rcpf(x); }
__device__ __forceinline__ float sigm(float x){ return fastrcp(1.0f + __expf(-x)); }

__device__ __forceinline__ v8f wmma16(v16h a, v16h b, v8f c){
  return __builtin_amdgcn_wmma_f32_16x16x32_f16(false, a, false, b, (short)0, c, false, false);
}

// A fragment: 16x32 f16 tile, row-major with leading dim ld (halves).
// lane<16: row=lane, K {0..15}; lane>=16: row=lane-16, K {16..31}
__device__ __forceinline__ v16h load_a_frag(const half_t* tile, int ld){
  int lane = threadIdx.x & 31;
  const half_t* p = tile + (lane & 15) * ld + ((lane >> 4) << 4);
  v16h a;
#pragma unroll
  for (int i = 0; i < 8; ++i) a[i] = p[i];
#pragma unroll
  for (int i = 0; i < 8; ++i) a[8 + i] = p[8 + i];
  return a;
}

// B fragment from B^T layout [N][K] row-major, leading dim ld:
// lane col = lane&15 ; 16 consecutive K at base (lane>=16 ? 16 : 0)
__device__ __forceinline__ v16h load_b_frag(const half_t* bt, int ld){
  int lane = threadIdx.x & 31;
  const half_t* p = bt + (lane & 15) * ld + ((lane >> 4) << 4);
  v16h b;
#pragma unroll
  for (int i = 0; i < 16; ++i) b[i] = p[i];
  return b;
}

// ---- CDNA5 async Global<->LDS bulk movers (ASYNCcnt-tracked) ----
__device__ __forceinline__ void async_g2l_b128(uint32_t lds_off, const void* gptr){
  uint64_t ga = (uint64_t)(uintptr_t)gptr;
  asm volatile("global_load_async_to_lds_b128 %0, %1, off"
               :: "v"(lds_off), "v"(ga) : "memory");
}
__device__ __forceinline__ void async_l2g_b128(void* gptr, uint32_t lds_off){
  uint64_t ga = (uint64_t)(uintptr_t)gptr;
  asm volatile("global_store_async_from_lds_b128 %0, %1, off"
               :: "v"(ga), "v"(lds_off) : "memory");
}
__device__ __forceinline__ void wait_async0(){
  asm volatile("s_wait_asynccnt 0x0" ::: "memory");
}

// ---------------- prep kernels ----------------

__global__ void k_transpose_f16(const float* W, half_t* outT, int K, int N){
  int i = blockIdx.x * blockDim.x + threadIdx.x;
  if (i >= K * N) return;
  int n = i / K, k = i % K;
  outT[n * K + k] = (half_t)W[k * N + n];
}

__global__ void k_copy_f32(const float* src, float* dst, int n){
  int i = blockIdx.x * blockDim.x + threadIdx.x;
  if (i < n) dst[i] = src[i];
}

__global__ void k_proj(const float* X, const float* W, const float* b, float* Y, int M){
  int i = blockIdx.x * blockDim.x + threadIdx.x;
  if (i >= M * 32) return;
  int r = i >> 5, n = i & 31;
  float acc = b[n];
#pragma unroll
  for (int k = 0; k < 32; ++k) acc += X[r * 32 + k] * W[k * 32 + n];
  Y[i] = acc;
}

__global__ void k_gather(const float* key, const float* ligp, const float* ligxyz,
                         float* ligg, float* lcg){
  int i = blockIdx.x * blockDim.x + threadIdx.x;
  if (i >= NB * NK * 35) return;
  int d = i % 35; int bk = i / 35; int b = bk / NK; int k = bk % NK;
  const float* kr = key + ((size_t)b * NK + k) * NLI;
  float acc = 0.f;
  if (d < 32){
    for (int j = 0; j < NLI; ++j) acc += kr[j] * ligp[((size_t)b * NLI + j) * 32 + d];
    ligg[((size_t)b * NK + k) * 32 + d] = acc;
  } else {
    int c = d - 32;
    for (int j = 0; j < NLI; ++j) acc += kr[j] * ligxyz[((size_t)b * NLI + j) * 3 + c];
    lcg[((size_t)b * NK + k) * 3 + c] = acc;
  }
}

__device__ __forceinline__ int bin_idx(float dx, float dy, float dz){
  float d2 = dx*dx + dy*dy + dz*dz;
  float dis = fminf(sqrtf(fmaxf(d2, 0.f)), 30.0f);
  int b = (int)floorf((dis + 1.0f) * 0.5f);
  return min(max(b, 0), 31);
}

__global__ void k_pair_rec(const float* xyz, int* idx){
  int i = blockIdx.x * blockDim.x + threadIdx.x;
  if (i >= NRE * NRE) return;
  int r = i / NRE, c = i % NRE;
  idx[i] = bin_idx(xyz[r*3]-xyz[c*3], xyz[r*3+1]-xyz[c*3+1], xyz[r*3+2]-xyz[c*3+2]);
}

__global__ void k_pair_lig(const float* lcg, int* idx){
  int i = blockIdx.x * blockDim.x + threadIdx.x;
  if (i >= NB * NK * NK) return;
  int jj = i % NK; int bi = i / NK; int ii = bi % NK; int b = bi / NK;
  const float* p = lcg + ((size_t)b * NK + ii) * 3;
  const float* q = lcg + ((size_t)b * NK + jj) * 3;
  idx[i] = bin_idx(p[0]-q[0], p[1]-q[1], p[2]-q[2]);
}

__global__ void k_make_z(const float* recp, const float* ligg, float* z){
  int i = blockIdx.x * blockDim.x + threadIdx.x;
  if (i >= NROW * 32) return;
  int d = i & 31; int row = i >> 5;
  int m = row % NK; int bn = row / NK; int n = bn % NRE; int b = bn / NRE;
  z[i] = recp[n * 32 + d] * ligg[((size_t)b * NK + m) * 32 + d];
}

// one-hot pair embedding tables: LN(onehot(t)) -> gated linears (32 entries)
__global__ void k_p2c_tables(const float* lnb, const float* lng,
    const float* g1w, const float* g1b, const float* l1w, const float* l1b,
    const float* g2w, const float* g2b, const float* l2w, const float* l2b,
    half_t* Ppp, half_t* Pcp){
  int t = threadIdx.x;
  if (t >= 32) return;
  float x[32];
  const float rs = rsqrtf(31.0f / 1024.0f + 1e-5f);   // var of one-hot(32)
#pragma unroll
  for (int d = 0; d < 32; ++d)
    x[d] = (((d == t) ? 1.0f : 0.0f) - 0.03125f) * rs * lng[d] + lnb[d];
  for (int c = 0; c < 32; ++c){
    float a1 = g1b[c], s1 = l1b[c], a2 = g2b[c], s2 = l2b[c];
    for (int d = 0; d < 32; ++d){
      float xv = x[d];
      a1 += xv * g1w[d*32+c]; s1 += xv * l1w[d*32+c];
      a2 += xv * g2w[d*32+c]; s2 += xv * l2w[d*32+c];
    }
    Pcp[t*32+c] = (half_t)(sigm(a1) * s1);   // gate1*lin1 -> cpair
    Ppp[t*32+c] = (half_t)(sigm(a2) * s2);   // gate2*lin2 -> ppair
  }
}

// ---------------- fused LN + WMMA stages ----------------

// LN(z) -> [gate1|lin1|gate2|lin2|egate] (32->160), gating epilogue
__global__ __launch_bounds__(256) void k_p2c_stage1(const float* z, const float* lnb,
    const float* lng, const half_t* WcatT, const float* bcat,
    half_t* ab1h, half_t* ab2h, half_t* gvec){
  __shared__ half_t zl[64*32];
  __shared__ float proj[64*160];
  int tid = threadIdx.x;
  size_t row0 = (size_t)blockIdx.x * 64;
  if (tid < 64){
    const float* zr = z + (row0 + tid) * 32;
    float mu = 0.f;
#pragma unroll
    for (int d = 0; d < 32; ++d) mu += zr[d];
    mu *= (1.f/32.f);
    float var = 0.f;
#pragma unroll
    for (int d = 0; d < 32; ++d){ float t = zr[d]-mu; var += t*t; }
    var *= (1.f/32.f);
    float rs = rsqrtf(var + 1e-5f);
#pragma unroll
    for (int d = 0; d < 32; ++d) zl[tid*32+d] = (half_t)((zr[d]-mu)*rs*lng[d] + lnb[d]);
  }
  __syncthreads();
  int wave = tid >> 5, lane = tid & 31;
  int colL = lane & 15, rb = (lane >> 4) << 3;
  for (int t = wave; t < 40; t += 8){
    int rt = t & 3, nt = t >> 2;
    v16h a  = load_a_frag(&zl[rt*16*32], 32);
    v16h bf = load_b_frag(WcatT + (nt*16)*32, 32);
    v8f acc = {};
    acc = wmma16(a, bf, acc);
#pragma unroll
    for (int r = 0; r < 8; ++r)
      proj[(rt*16 + rb + r)*160 + nt*16 + colL] = acc[r] + bcat[nt*16 + colL];
  }
  __syncthreads();
  for (int e = tid; e < 64*32; e += 256){
    int r = e >> 5, c = e & 31;
    const float* pr = &proj[r*160];
    size_t gi = (row0 + r)*32 + c;
    ab1h[gi] = (half_t)(sigm(pr[c])      * pr[32+c]);
    ab2h[gi] = (half_t)(sigm(pr[64+c])   * pr[96+c]);
    gvec[gi] = (half_t)(sigm(pr[128+c]));
  }
}

// block1 (K=400 via rec-pair table) + block2 (K=64 via lig-pair table), one wave/tile
__global__ __launch_bounds__(32) void k_p2c_block(const half_t* PppG, const half_t* PcpG,
    const int* ridx, const int* lidx, const half_t* ab1h, const half_t* ab2h, float* bsum){
  __shared__ half_t At[16*32], Bt[16*32];
  __shared__ half_t Pl[1024], Pc[1024];
  int lane = threadIdx.x;
  int i0 = blockIdx.x * 16, j0 = blockIdx.y * 16;
  int b = blockIdx.z >> 5, c = blockIdx.z & 31;
  for (int e = lane; e < 1024; e += 32){ Pl[e] = PppG[e]; Pc[e] = PcpG[e]; }
  __syncthreads();
  v8f acc = {};
  for (int k0 = 0; k0 < NRE; k0 += 32){
    for (int e = lane; e < 512; e += 32){
      int ii = e >> 5, kk = e & 31, kg = k0 + kk;
      half_t av = (half_t)0.f, bv = (half_t)0.f;
      if (kg < NRE){
        av = Pl[ridx[(i0+ii)*NRE + kg]*32 + c];
        bv = ab1h[(((size_t)b*NRE + kg)*NK + j0 + ii)*32 + c];
      }
      At[e] = av; Bt[e] = bv;
    }
    if (k0 + 32 < NRE)
      __builtin_prefetch(&ab1h[(((size_t)b*NRE + k0+32)*NK + j0)*32 + c], 0, 1);
    __syncthreads();
    v16h a  = load_a_frag(At, 32);
    v16h bf = load_b_frag(Bt, 32);
    acc = wmma16(a, bf, acc);
    __syncthreads();
  }
  for (int k0 = 0; k0 < NK; k0 += 32){
    for (int e = lane; e < 512; e += 32){
      int ii = e >> 5, kk = e & 31, kg = k0 + kk;
      At[e] = ab2h[(((size_t)b*NRE + i0 + ii)*NK + kg)*32 + c];
      Bt[e] = Pc[lidx[((size_t)b*NK + j0 + ii)*NK + kg]*32 + c];
    }
    __syncthreads();
    v16h a  = load_a_frag(At, 32);
    v16h bf = load_b_frag(Bt, 32);
    acc = wmma16(a, bf, acc);
    __syncthreads();
  }
  int colL = lane & 15, rb = (lane >> 4) << 3;
#pragma unroll
  for (int r = 0; r < 8; ++r)
    bsum[(((size_t)b*NRE + i0 + rb + r)*NK + j0 + colL)*32 + c] = acc[r];
}

// z += g * (LN_c(block1+block2) @ after + b)
__global__ __launch_bounds__(256) void k_p2c_final(float* z, const float* bsum,
    const float* lncb, const float* lncg, const half_t* afterT, const float* afterb,
    const half_t* gvec){
  __shared__ half_t th[64*32];
  int tid = threadIdx.x;
  size_t row0 = (size_t)blockIdx.x * 64;
  if (tid < 64){
    const float* br = bsum + (row0 + tid) * 32;
    float mu = 0.f;
#pragma unroll
    for (int d = 0; d < 32; ++d) mu += br[d];
    mu *= (1.f/32.f);
    float var = 0.f;
#pragma unroll
    for (int d = 0; d < 32; ++d){ float t = br[d]-mu; var += t*t; }
    var *= (1.f/32.f);
    float rs = rsqrtf(var + 1e-5f);
#pragma unroll
    for (int d = 0; d < 32; ++d) th[tid*32+d] = (half_t)((br[d]-mu)*rs*lncg[d] + lncb[d]);
  }
  __syncthreads();
  int wave = tid >> 5, lane = tid & 31, colL = lane & 15, rb = (lane >> 4) << 3;
  int rt = wave & 3, nt = wave >> 2;
  v16h a  = load_a_frag(&th[rt*16*32], 32);
  v16h bf = load_b_frag(afterT + nt*16*32, 32);
  v8f acc = {};
  acc = wmma16(a, bf, acc);
#pragma unroll
  for (int r = 0; r < 8; ++r){
    size_t gi = (row0 + rt*16 + rb + r)*32 + nt*16 + colL;
    z[gi] += (float)gvec[gi] * (acc[r] + afterb[nt*16 + colL]);
  }
}

// fully fused row-wise triangle attention for one (b, receptor-row): 64 ligand keys
__global__ __launch_bounds__(256) void k_attn(float* z, const float* lnb, const float* lng,
    const half_t* WqT, const half_t* WkT, const half_t* WvT, const half_t* WgT,
    const float* gb, const half_t* finT, const float* finb){
  __shared__ half_t zl[64*32];
  __shared__ half_t qh[64*32], kh[64*32], vh[64*32], gh[64*32];
  __shared__ __align__(16) char pool[64*64*4];  // logits f32, later aliased by v^T f16
  __shared__ half_t ph[64*64];
  __shared__ half_t og[64*128];
  float*  lg = (float*)pool;
  half_t* vT = (half_t*)pool;

  int tid = threadIdx.x, wave = tid >> 5, lane = tid & 31;
  int colL = lane & 15, rb = (lane >> 4) << 3;
  size_t base = (size_t)blockIdx.x * 64 * 32;

  if (tid < 64){
    const float* zr = z + base + (size_t)tid * 32;
    float mu = 0.f;
#pragma unroll
    for (int d = 0; d < 32; ++d) mu += zr[d];
    mu *= (1.f/32.f);
    float var = 0.f;
#pragma unroll
    for (int d = 0; d < 32; ++d){ float t = zr[d]-mu; var += t*t; }
    var *= (1.f/32.f);
    float rs = rsqrtf(var + 1e-5f);
#pragma unroll
    for (int d = 0; d < 32; ++d) zl[tid*32+d] = (half_t)((zr[d]-mu)*rs*lng[d] + lnb[d]);
  }
  __syncthreads();

  const float scale = 0.17677669529663687f;  // 32^-0.5

  for (int h = 0; h < 4; ++h){
    // q/k/v/g head projections: 32 tiles over 8 waves
    for (int t = wave; t < 32; t += 8){
      int mat = t >> 3, rt = t & 3, nt = (t >> 2) & 1;
      const half_t* WT = (mat==0) ? WqT : (mat==1) ? WkT : (mat==2) ? WvT : WgT;
      v16h a  = load_a_frag(&zl[rt*16*32], 32);
      v16h bf = load_b_frag(WT + (h*32 + nt*16)*32, 32);
      v8f acc = {};
      acc = wmma16(a, bf, acc);
      half_t* dst = (mat==0) ? qh : (mat==1) ? kh : (mat==2) ? vh : gh;
#pragma unroll
      for (int r = 0; r < 8; ++r){
        int row = rt*16 + rb + r, col = nt*16 + colL;
        float v = acc[r];
        if (mat == 0) v *= scale;
        if (mat == 3) v = sigm(v + gb[h*32 + col]);
        dst[row*32 + col] = (half_t)v;
      }
    }
    __syncthreads();
    // logits = q @ k^T (K=32, single wmma per 16x16 tile)
    for (int t = wave; t < 16; t += 8){
      int qt = t & 3, kt = t >> 2;
      v16h a  = load_a_frag(&qh[qt*16*32], 32);
      v16h bf = load_b_frag(&kh[kt*16*32], 32);
      v8f acc = {};
      acc = wmma16(a, bf, acc);
#pragma unroll
      for (int r = 0; r < 8; ++r) lg[(qt*16 + rb + r)*64 + kt*16 + colL] = acc[r];
    }
    __syncthreads();
    // softmax rows (mask is all-ones -> no bias)
    if (tid < 64){
      float* Lr = &lg[tid*64];
      float mx = -3.0e38f;
      for (int j = 0; j < 64; ++j) mx = fmaxf(mx, Lr[j]);
      float s = 0.f;
      for (int j = 0; j < 64; ++j){ float ev = __expf(Lr[j]-mx); s += ev; ph[tid*64+j] = (half_t)ev; }
      float inv = fastrcp(s);
      for (int j = 0; j < 64; ++j) ph[tid*64+j] = (half_t)((float)ph[tid*64+j] * inv);
    }
    __syncthreads();                         // lg dead -> pool reused as vT
    for (int e = tid; e < 2048; e += 256){
      int cI = e >> 6, kI = e & 63;
      vT[cI*64 + kI] = vh[kI*32 + cI];
    }
    __syncthreads();
    // o = softmax @ v, gated: one tile per wave (K=64 -> 2 wmma)
    {
      int rt = wave & 3, nt = wave >> 2;
      v8f acc = {};
#pragma unroll
      for (int kc = 0; kc < 2; ++kc){
        v16h a  = load_a_frag(&ph[rt*16*64 + kc*32], 64);
        v16h bf = load_b_frag(&vT[(nt*16)*64 + kc*32], 64);
        acc = wmma16(a, bf, acc);
      }
#pragma unroll
      for (int r = 0; r < 8; ++r){
        int row = rt*16 + rb + r, col = nt*16 + colL;
        og[row*128 + h*32 + col] = (half_t)(acc[r] * (float)gh[row*32 + col]);
      }
    }
    __syncthreads();
  }
  // fin: (64x128)@(128x32) + bias, residual into z
  {
    int rt = wave & 3, nt = wave >> 2;
    v8f acc = {};
#pragma unroll
    for (int kc = 0; kc < 4; ++kc){
      v16h a  = load_a_frag(&og[rt*16*128 + kc*32], 128);
      v16h bf = load_b_frag(finT + (nt*16)*128 + kc*32, 128);
      acc = wmma16(a, bf, acc);
    }
#pragma unroll
    for (int r = 0; r < 8; ++r){
      size_t gi = base + (size_t)(rt*16 + rb + r)*32 + nt*16 + colL;
      z[gi] += acc[r] + finb[nt*16 + colL];
    }
  }
}

// z = relu(LN(z) @ l1 + b1) @ l2 + b2  (no residual)
__global__ __launch_bounds__(256) void k_transition(float* z, const float* lnb,
    const float* lng, const half_t* l1T, const float* b1, const half_t* l2T,
    const float* b2){
  __shared__ half_t zl[64*32];
  __shared__ half_t h1[64*128];
  int tid = threadIdx.x, wave = tid >> 5, lane = tid & 31;
  int colL = lane & 15, rb = (lane >> 4) << 3;
  size_t row0 = (size_t)blockIdx.x * 64;
  if (tid < 64){
    const float* zr = z + (row0 + tid) * 32;
    float mu = 0.f;
#pragma unroll
    for (int d = 0; d < 32; ++d) mu += zr[d];
    mu *= (1.f/32.f);
    float var = 0.f;
#pragma unroll
    for (int d = 0; d < 32; ++d){ float t = zr[d]-mu; var += t*t; }
    var *= (1.f/32.f);
    float rs = rsqrtf(var + 1e-5f);
#pragma unroll
    for (int d = 0; d < 32; ++d) zl[tid*32+d] = (half_t)((zr[d]-mu)*rs*lng[d] + lnb[d]);
  }
  __syncthreads();
  for (int t = wave; t < 32; t += 8){
    int rt = t & 3, nt = t >> 2;
    v16h a  = load_a_frag(&zl[rt*16*32], 32);
    v16h bf = load_b_frag(l1T + nt*16*32, 32);
    v8f acc = {};
    acc = wmma16(a, bf, acc);
#pragma unroll
    for (int r = 0; r < 8; ++r){
      int row = rt*16 + rb + r, col = nt*16 + colL;
      h1[row*128 + col] = (half_t)fmaxf(acc[r] + b1[col], 0.f);
    }
  }
  __syncthreads();
  {
    int rt = wave & 3, nt = wave >> 2;
    v8f acc = {};
#pragma unroll
    for (int kc = 0; kc < 4; ++kc){
      v16h a  = load_a_frag(&h1[rt*16*128 + kc*32], 128);
      v16h bf = load_b_frag(l2T + (nt*16)*128 + kc*32, 128);
      acc = wmma16(a, bf, acc);
    }
#pragma unroll
    for (int r = 0; r < 8; ++r){
      size_t gi = (row0 + rt*16 + rb + r)*32 + nt*16 + colL;
      z[gi] = acc[r] + b2[nt*16 + colL];
    }
  }
}

// bulk z -> out copy through LDS using CDNA5 async global<->LDS movers.
// grid exactly covers NROW*32 elements (4 floats / 16B per thread, full EXEC).
__global__ __launch_bounds__(256) void k_out_async(const float* z, float* out){
  __shared__ __align__(16) float buf[256 * 4];
  int tid = threadIdx.x;
  size_t i = ((size_t)blockIdx.x * 256 + tid) * 4;
  uint32_t lds = (uint32_t)(tid * 16);
  (void)buf;
  async_g2l_b128(lds, z + i);
  wait_async0();
  async_l2g_b128(out + i, lds);
  wait_async0();
}

// masks (all ones) + gathered ligand features tail
__global__ void k_out_tail(const float* ligg, float* out){
  const size_t NZ  = (size_t)NROW * 32;
  const int NM  = NROW;
  const int NHg = NB * NK * 32;
  const int total = NM + NHg + NB * NK;
  int i = blockIdx.x * blockDim.x + threadIdx.x;
  if (i >= total) return;
  float v;
  if (i < NM)            v = 1.0f;
  else if (i < NM + NHg) v = ligg[i - NM];
  else                   v = 1.0f;
  out[NZ + i] = v;
}

// ---------------- host ----------------

extern "C" void kernel_launch(void* const* d_in, const int* in_sizes, int n_in,
                              void* d_out, int out_size, void* d_ws, size_t ws_size,
                              hipStream_t stream){
  (void)in_sizes; (void)n_in; (void)out_size; (void)ws_size;
  const float* hs_rec = (const float*)d_in[0];
  const float* hs_lig = (const float*)d_in[1];
  const float* recxyz = (const float*)d_in[2];
  const float* ligxyz = (const float*)d_in[3];
  const float* key    = (const float*)d_in[4];
  auto F = [&](int i){ return (const float*)d_in[i]; };
  // params in JAX pytree (sorted-key) order after the 5 tensor inputs:
  // 5 Wls.b 6 Wls.w 7 Wrs.b 8 Wrs.w
  // attn[s] at 9/18: fin.b fin.w g.b g.w ln.b ln.g wk wq wv
  // p2c[s] at 27/43: after.b after.w egate.b egate.w gate1.b gate1.w gate2.b
  //                  gate2.w lin1.b lin1.w lin2.b lin2.w ln.b ln.g ln_c.b ln_c.g
  // trans at 59: l1.b l1.w l2.b l2.w ln.b ln.g
  const int IA[2] = {9, 18};
  const int IP[2] = {27, 43};

  char* cur = (char*)d_ws;
  auto carve = [&](size_t bytes)->char*{
    char* p = cur; cur += (bytes + 255) & ~((size_t)255); return p;
  };
  float*  z    = (float*)carve((size_t)NROW * 32 * 4);
  float*  bsum = (float*)carve((size_t)NROW * 32 * 4);
  half_t* ab1h = (half_t*)carve((size_t)NROW * 32 * 2);
  half_t* ab2h = (half_t*)carve((size_t)NROW * 32 * 2);
  half_t* gvec = (half_t*)carve((size_t)NROW * 32 * 2);
  int*    ridx = (int*)carve((size_t)NRE * NRE * 4);
  int*    lidx = (int*)carve((size_t)NB * NK * NK * 4);
  float*  recp = (float*)carve((size_t)NRE * 32 * 4);
  float*  ligp = (float*)carve((size_t)NB * NLI * 32 * 4);
  float*  ligg = (float*)carve((size_t)NB * NK * 32 * 4);
  float*  lcg  = (float*)carve((size_t)NB * NK * 3 * 4);
  half_t *Ppp[2], *Pcp[2], *WcatT[2], *afterT[2], *WqT[2], *WkT[2], *WvT[2], *WgT[2], *finT[2];
  float* bcat[2];
  for (int s = 0; s < 2; ++s){
    Ppp[s]   = (half_t*)carve(1024 * 2);
    Pcp[s]   = (half_t*)carve(1024 * 2);
    WcatT[s] = (half_t*)carve(160 * 32 * 2);
    bcat[s]  = (float*)carve(160 * 4);
    afterT[s]= (half_t*)carve(1024 * 2);
    WqT[s]   = (half_t*)carve(128 * 32 * 2);
    WkT[s]   = (half_t*)carve(128 * 32 * 2);
    WvT[s]   = (half_t*)carve(128 * 32 * 2);
    WgT[s]   = (half_t*)carve(128 * 32 * 2);
    finT[s]  = (half_t*)carve(32 * 128 * 2);
  }
  half_t* l1T = (half_t*)carve(128 * 32 * 2);
  half_t* l2T = (half_t*)carve(32 * 128 * 2);

  auto tp = [&](const float* W, half_t* T, int K, int N){
    int n = K * N;
    k_transpose_f16<<<dim3((n + 255) / 256), dim3(256), 0, stream>>>(W, T, K, N);
  };
  auto cp = [&](const float* src, float* dst, int n){
    k_copy_f32<<<dim3((n + 255) / 256), dim3(256), 0, stream>>>(src, dst, n);
  };

  for (int s = 0; s < 2; ++s){
    int p = IP[s], a = IA[s];
    tp(F(p + 5),  WcatT[s] + 0,    32, 32);   // gate1.w
    tp(F(p + 9),  WcatT[s] + 1024, 32, 32);   // lin1.w
    tp(F(p + 7),  WcatT[s] + 2048, 32, 32);   // gate2.w
    tp(F(p + 11), WcatT[s] + 3072, 32, 32);   // lin2.w
    tp(F(p + 3),  WcatT[s] + 4096, 32, 32);   // egate.w
    tp(F(p + 1),  afterT[s], 32, 32);         // after.w
    cp(F(p + 4),  bcat[s] + 0,   32);         // gate1.b
    cp(F(p + 8),  bcat[s] + 32,  32);         // lin1.b
    cp(F(p + 6),  bcat[s] + 64,  32);         // gate2.b
    cp(F(p + 10), bcat[s] + 96,  32);         // lin2.b
    cp(F(p + 2),  bcat[s] + 128, 32);         // egate.b
    tp(F(a + 7), WqT[s], 32, 128);            // wq
    tp(F(a + 6), WkT[s], 32, 128);            // wk
    tp(F(a + 8), WvT[s], 32, 128);            // wv
    tp(F(a + 3), WgT[s], 32, 128);            // g.w
    tp(F(a + 1), finT[s], 128, 32);           // fin.w
    k_p2c_tables<<<dim3(1), dim3(32), 0, stream>>>(
        F(p+12), F(p+13), F(p+5), F(p+4), F(p+9), F(p+8),
        F(p+7), F(p+6), F(p+11), F(p+10), Ppp[s], Pcp[s]);
  }
  tp(F(60), l1T, 32, 128);
  tp(F(62), l2T, 128, 32);

  k_proj<<<dim3((NRE*32 + 255)/256), dim3(256), 0, stream>>>(hs_rec, F(8), F(7), recp, NRE);
  k_proj<<<dim3((NB*NLI*32 + 255)/256), dim3(256), 0, stream>>>(hs_lig, F(6), F(5), ligp, NB*NLI);
  k_gather<<<dim3((NB*NK*35 + 255)/256), dim3(256), 0, stream>>>(key, ligp, ligxyz, ligg, lcg);
  k_pair_rec<<<dim3((NRE*NRE + 255)/256), dim3(256), 0, stream>>>(recxyz, ridx);
  k_pair_lig<<<dim3((NB*NK*NK + 255)/256), dim3(256), 0, stream>>>(lcg, lidx);
  k_make_z<<<dim3((NROW*32 + 255)/256), dim3(256), 0, stream>>>(recp, ligg, z);

  for (int s = 0; s < 2; ++s){
    int p = IP[s], a = IA[s];
    k_p2c_stage1<<<dim3(NROW/64), dim3(256), 0, stream>>>(
        z, F(p+12), F(p+13), WcatT[s], bcat[s], ab1h, ab2h, gvec);
    k_p2c_block<<<dim3(NRE/16, NK/16, NB*32), dim3(32), 0, stream>>>(
        Ppp[s], Pcp[s], ridx, lidx, ab1h, ab2h, bsum);
    k_p2c_final<<<dim3(NROW/64), dim3(256), 0, stream>>>(
        z, bsum, F(p+14), F(p+15), afterT[s], F(p+0), gvec);
    k_attn<<<dim3(NB*NRE), dim3(256), 0, stream>>>(
        z, F(a+4), F(a+5), WqT[s], WkT[s], WvT[s], WgT[s], F(a+2), finT[s], F(a+0));
    k_transition<<<dim3(NROW/64), dim3(256), 0, stream>>>(
        z, F(63), F(64), l1T, F(59), l2T, F(61));
  }

  // NROW*32 = 6,553,600 floats -> 1,638,400 threads of 4 floats = 6400 blocks exactly
  k_out_async<<<dim3((NROW*32) / (256*4)), dim3(256), 0, stream>>>(z, (float*)d_out);
  int tail = NROW + NB*NK*32 + NB*NK;
  k_out_tail<<<dim3((tail + 255)/256), dim3(256), 0, stream>>>(ligg, (float*)d_out);
}
